// CISTransformerDecoder_43336220016864
// MI455X (gfx1250) — matrix operand order repaired and checked by reading, not compiled, not run on hardware
//
#include <hip/hip_runtime.h>
#include <hip/hip_bf16.h>
#include <cstdint>
#include <cstddef>

// ---------------- problem constants ----------------
#define NB      2
#define DMODEL  256
#define NHEAD   8
#define NLVL    4
#define NPTS    4
#define NLAYERS 6
#define NFFN    1024
#define DHEAD   32
#define LQ      5440          // 64*64+32*32+16*16+8*8
#define NTROWS  (NB * LQ)     // 10880

// ---------------- WMMA types ----------------
typedef __attribute__((ext_vector_type(16))) __bf16 v16bf;
typedef __attribute__((ext_vector_type(8)))  float  v8f;

// hardware RNE f32->bf16 (v_cvt_*_bf16_f32), packed into a dword
__device__ __forceinline__ unsigned int pack_bf16x2(float lo, float hi) {
  __bf16 l = (__bf16)lo, h = (__bf16)hi;
  unsigned short lb = __builtin_bit_cast(unsigned short, l);
  unsigned short hb = __builtin_bit_cast(unsigned short, h);
  return (unsigned int)lb | ((unsigned int)hb << 16);
}

__device__ __forceinline__ void qlevel(int q, int& l, int& hw, int& st) {
  if (q < 4096)      { l = 0; hw = 64; st = 0;    }
  else if (q < 5120) { l = 1; hw = 32; st = 4096; }
  else if (q < 5376) { l = 2; hw = 16; st = 5120; }
  else               { l = 3; hw = 8;  st = 5376; }
}

// ---------------- prep: flatten + level embed + val=mem+posm+le ----------------
__global__ __launch_bounds__(256) void cis_prep_kernel(
    const float* __restrict__ s0, const float* __restrict__ s1,
    const float* __restrict__ s2, const float* __restrict__ s3,
    const float* __restrict__ p0, const float* __restrict__ p1,
    const float* __restrict__ p2, const float* __restrict__ p3,
    const float* __restrict__ m0, const float* __restrict__ m1,
    const float* __restrict__ m2, const float* __restrict__ m3,
    const float* __restrict__ q0, const float* __restrict__ q1,
    const float* __restrict__ q2, const float* __restrict__ q3,
    const float* __restrict__ le,
    float* __restrict__ outb, float* __restrict__ posb, float* __restrict__ valb) {
  size_t t  = (size_t)blockIdx.x * 256 + threadIdx.x;   // exactly NTROWS*256 threads
  int d  = (int)(t & 255);
  int nt = (int)(t >> 8);
  int b  = nt / LQ;
  int q  = nt - b * LQ;
  int l, hw, st;
  qlevel(q, l, hw, st);
  int local = q - st;
  int y = local / hw;
  int x = local - y * hw;
  const float *sp, *pp, *mp, *qp;
  if (l == 0)      { sp = s0; pp = p0; mp = m0; qp = q0; }
  else if (l == 1) { sp = s1; pp = p1; mp = m1; qp = q1; }
  else if (l == 2) { sp = s2; pp = p2; mp = m2; qp = q2; }
  else             { sp = s3; pp = p3; mp = m3; qp = q3; }
  size_t off = (((size_t)b * DMODEL + d) * hw + y) * hw + x;   // [B,D,H,W]
  float lev = le[l * DMODEL + d];
  outb[t] = sp[off];
  posb[t] = pp[off] + lev;
  valb[t] = mp[off] + qp[off] + lev;
}

// ---------------- bf16 WMMA GEMM: C = act((A[+A2]) * W + bias) ----------------
// A: [Mrows,K] f32 row-major, W: [K,N] f32 row-major, C: [Mrows,N] f32.
// Block tile 128x128, 8 waves (2 along M x 4 along N), wave tile 64x32:
// 8x v_wmma_f32_16x16x32_bf16 per 32-deep k-step, fragments via ds_load_b128.
#define BM 128
#define BN 128
#define BK 32
#define A_STRIDE 40   // ushorts per LDS row (32 data + 8 pad -> 80B: bank-safe, 16B aligned)
#define B_STRIDE 40

__global__ __launch_bounds__(256) void cis_gemm_bf16_wmma(
    const float* __restrict__ A, const float* __restrict__ A2,
    const float* __restrict__ W, const float* __restrict__ bias,
    float* __restrict__ C, int Mrows, int N, int K, int do_relu) {
  __shared__ __align__(16) unsigned short As[BM * A_STRIDE];   // A tile  [row][k]
  __shared__ __align__(16) unsigned short Bs[BN * B_STRIDE];   // W tile transposed [col][k]

  const int tid  = threadIdx.x;
  const int wave = tid >> 5;
  const int lane = tid & 31;
  const int block_m = blockIdx.x * BM;
  const int block_n = blockIdx.y * BN;
  const int wm = (wave & 1) * 64;    // wave row offset within block tile (2 waves on M)
  const int wn = (wave >> 1) * 32;   // wave col offset (4 waves on N)

  // ISA 16-bit A-matrix lane layout (16x32): lanes 0-15 row=lane, K base 0; lanes 16-31 K base 8
  const int arow  = lane & 15;
  const int abase = (lane >> 4) * 8;
  // ISA 16-bit B-matrix lane layout (32x16): col = lane&15, K base = (lane>>4)*8 (same as A)
  const int bcol = lane & 15;
  const int bkb  = (lane >> 4) * 8;

  v8f acc[4][2] = {};

  // staging indices
  const int ar = tid >> 1;            // A: row 0..127
  const int ac = (tid & 1) * 16;      // A: col 0 or 16
  const int bc = tid & 127;           // B: column 0..127
  const int bk0 = (tid >> 7) * 16;    // B: k 0 or 16

  for (int kk = 0; kk < K; kk += BK) {
    // ---- stage A tile (BMxBK f32 -> bf16): 16 contiguous floats/thread, 2x ds_store_b128 ----
    {
      const float* ap = A + (size_t)(block_m + ar) * K + kk + ac;
      float f[16];
      if (A2) {
        const float* ap2 = A2 + (size_t)(block_m + ar) * K + kk + ac;
#pragma unroll
        for (int i = 0; i < 16; ++i) f[i] = ap[i] + ap2[i];
      } else {
#pragma unroll
        for (int i = 0; i < 16; ++i) f[i] = ap[i];
      }
      unsigned int u[8];
#pragma unroll
      for (int j = 0; j < 8; ++j) u[j] = pack_bf16x2(f[2 * j], f[2 * j + 1]);
      uint4* dst = (uint4*)&As[ar * A_STRIDE + ac];
      dst[0] = make_uint4(u[0], u[1], u[2], u[3]);
      dst[1] = make_uint4(u[4], u[5], u[6], u[7]);
    }
    // ---- stage W tile transposed (BKxBN -> Bs[col][k]): thread reads one column over 16 k
    //      (per-instruction the wave reads 32 contiguous floats of a W row: coalesced) ----
    {
      const float* wp = W + (size_t)(kk + bk0) * N + block_n + bc;
      float f[16];
#pragma unroll
      for (int i = 0; i < 16; ++i) f[i] = wp[(size_t)i * N];
      unsigned int u[8];
#pragma unroll
      for (int j = 0; j < 8; ++j) u[j] = pack_bf16x2(f[2 * j], f[2 * j + 1]);
      uint4* dst = (uint4*)&Bs[bc * B_STRIDE + bk0];
      dst[0] = make_uint4(u[0], u[1], u[2], u[3]);
      dst[1] = make_uint4(u[4], u[5], u[6], u[7]);
    }
    __syncthreads();

    // ---- assemble fragments (2x ds_load_b128 each) and issue 8 WMMAs ----
    union Frag { uint4 q[2]; v16bf v; };
    Frag a[4], b[2];
#pragma unroll
    for (int i = 0; i < 4; ++i) {
      const unsigned short* arp = &As[(wm + 16 * i + arow) * A_STRIDE];
      a[i].q[0] = *(const uint4*)(arp + abase);
      a[i].q[1] = *(const uint4*)(arp + abase + 16);
    }
#pragma unroll
    for (int j = 0; j < 2; ++j) {
      const unsigned short* brp = &Bs[(wn + 16 * j + bcol) * B_STRIDE];
      b[j].q[0] = *(const uint4*)(brp + bkb);
      b[j].q[1] = *(const uint4*)(brp + bkb + 16);
    }
#pragma unroll
    for (int i = 0; i < 4; ++i) {
#pragma unroll
      for (int j = 0; j < 2; ++j) {
        acc[i][j] = __builtin_amdgcn_wmma_f32_16x16x32_bf16(
            false, a[i].v, false, b[j].v, (short)0, acc[i][j], false, false);
      }
    }
    __syncthreads();
  }

  // ---- epilogue: ISA 16x16 f32 D layout: lane 0-15 N=lane M=r; lane16-31 M=r+8 ----
  const int crow = (lane >> 4) * 8;
  const int ccol = lane & 15;
#pragma unroll
  for (int i = 0; i < 4; ++i) {
#pragma unroll
    for (int j = 0; j < 2; ++j) {
      int gm = block_m + wm + 16 * i + crow;
      int gn = block_n + wn + 16 * j + ccol;
      float bi = bias ? bias[gn] : 0.0f;
#pragma unroll
      for (int r = 0; r < 8; ++r) {
        float v = acc[i][j][r] + bi;
        if (do_relu) v = v > 0.0f ? v : 0.0f;
        C[(size_t)(gm + r) * N + gn] = v;
      }
    }
  }
}

// ---------------- softmax over 16 attention weights per (b,q,head) ----------------
__global__ __launch_bounds__(256) void cis_softmax16(float* __restrict__ aw, int rows) {
  int r = blockIdx.x * 256 + threadIdx.x;
  if (r >= rows) return;
  float* p = aw + (size_t)r * 16;   // [NT,128] with head-major inner: row stride 16 works
  float mx = p[0];
#pragma unroll
  for (int i = 1; i < 16; ++i) mx = fmaxf(mx, p[i]);
  float s = 0.0f;
  float e[16];
#pragma unroll
  for (int i = 0; i < 16; ++i) { e[i] = __expf(p[i] - mx); s += e[i]; }
  float inv = 1.0f / s;
#pragma unroll
  for (int i = 0; i < 16; ++i) p[i] = e[i] * inv;
}

// ---------------- MSDA bilinear sampling: wave per (b,q,head), lane = channel ----------------
__global__ __launch_bounds__(256) void cis_msda_sample(
    const float* __restrict__ v, const float* __restrict__ off,
    const float* __restrict__ aw, float* __restrict__ outb) {
  const int row  = blockIdx.x * 8 + (threadIdx.x >> 5);   // over NT*NHEAD
  const int lane = threadIdx.x & 31;                      // channel within head
  const int m  = row & 7;
  const int nt = row >> 3;
  const int b  = nt / LQ;
  const int q  = nt - b * LQ;
  int ql, qhw, qst;
  qlevel(q, ql, qhw, qst);
  int local = q - qst;
  int qy = local / qhw, qx = local - qy * qhw;
  const float refx = (qx + 0.5f) / (float)qhw;
  const float refy = (qy + 0.5f) / (float)qhw;

  const int dims[NLVL]   = {64, 32, 16, 8};
  const int starts[NLVL] = {0, 4096, 5120, 5376};

  const float* offp = off + (size_t)nt * 256 + m * 32;    // [M][LVL][PTS][2]
  const float* awp  = aw  + (size_t)nt * 128 + m * 16;    // [M][LVL*PTS]
  const float* vb   = v   + (size_t)b * LQ * DMODEL;

  float acc = 0.0f;
  for (int l = 0; l < NLVL; ++l) {
    const int hw = dims[l], st = starts[l];
    const float fhw = (float)hw;
#pragma unroll
    for (int p = 0; p < NPTS; ++p) {
      float ox = offp[l * 8 + p * 2 + 0];
      float oy = offp[l * 8 + p * 2 + 1];
      float a  = awp[l * 4 + p];
      // loc = ref + off/(w,h);  x = loc_x*w - 0.5  ==  ref_x*w + off_x - 0.5
      float x = refx * fhw + ox - 0.5f;
      float y = refy * fhw + oy - 0.5f;
      float x0f = floorf(x), y0f = floorf(y);
      float wx1 = x - x0f, wy1 = y - y0f;
      int x0 = (int)x0f, y0 = (int)y0f;
#pragma unroll
      for (int c = 0; c < 4; ++c) {
        int dx = c & 1, dy = c >> 1;
        int xi = x0 + dx, yi = y0 + dy;
        float wx = dx ? wx1 : (1.0f - wx1);
        float wy = dy ? wy1 : (1.0f - wy1);
        if (xi >= 0 && xi < hw && yi >= 0 && yi < hw) {
          acc += a * wx * wy *
                 vb[(size_t)(st + yi * hw + xi) * DMODEL + m * 32 + lane];
        }
      }
    }
  }
  outb[(size_t)nt * DMODEL + m * 32 + lane] = acc;
}

// ---------------- residual + layernorm: wave per row (D=256, 8 per lane) ----------------
__global__ __launch_bounds__(256) void cis_add_ln(
    const float* __restrict__ x, const float* __restrict__ a,
    const float* __restrict__ g, const float* __restrict__ bta,
    float* __restrict__ out, int rows) {
  int row  = blockIdx.x * 8 + (threadIdx.x >> 5);
  int lane = threadIdx.x & 31;
  if (row >= rows) return;
  const float* xr = x + (size_t)row * DMODEL;
  const float* ar = a + (size_t)row * DMODEL;
  float v[8], s = 0.0f;
#pragma unroll
  for (int i = 0; i < 8; ++i) { v[i] = xr[lane + 32 * i] + ar[lane + 32 * i]; s += v[i]; }
#pragma unroll
  for (int o = 16; o > 0; o >>= 1) s += __shfl_xor(s, o, 32);
  float mean = s * (1.0f / DMODEL);
  float var = 0.0f;
#pragma unroll
  for (int i = 0; i < 8; ++i) { float d = v[i] - mean; var += d * d; }
#pragma unroll
  for (int o = 16; o > 0; o >>= 1) var += __shfl_xor(var, o, 32);
  float rinv = rsqrtf(var * (1.0f / DMODEL) + 1e-5f);
  float* orow = out + (size_t)row * DMODEL;
#pragma unroll
  for (int i = 0; i < 8; ++i)
    orow[lane + 32 * i] = (v[i] - mean) * rinv * g[lane + 32 * i] + bta[lane + 32 * i];
}

// ---------------- finalize: copy result + level_start_index ints ----------------
__global__ __launch_bounds__(256) void cis_finalize(const float* __restrict__ src,
                                                    float* __restrict__ dst) {
  size_t i = (size_t)blockIdx.x * 256 + threadIdx.x;
  const size_t n = (size_t)NTROWS * DMODEL;
  if (i < n) {
    dst[i] = src[i];
  } else if (i < n + 4) {
    const int lsi[4] = {0, 4096, 5120, 5376};
    ((int*)dst)[i] = lsi[i - n];
  }
}

// ---------------- host orchestration ----------------
extern "C" void kernel_launch(void* const* d_in, const int* in_sizes, int n_in,
                              void* d_out, int out_size, void* d_ws, size_t ws_size,
                              hipStream_t stream) {
  (void)in_sizes; (void)n_in; (void)out_size; (void)ws_size;
  auto f = [&](int i) { return (const float*)d_in[i]; };

  // features: src,pos,mem,posm per level (setup_inputs insertion order)
  const float *src[4], *pos[4], *mem[4], *posm[4];
  for (int l = 0; l < 4; ++l) {
    src[l]  = f(l * 4 + 0);
    pos[l]  = f(l * 4 + 1);
    mem[l]  = f(l * 4 + 2);
    posm[l] = f(l * 4 + 3);
  }
  // params pytree, JAX leaf order: layers[0..5] (sorted keys), then level_embed
  const int PB = 16;
  const float* le = f(PB + NLAYERS * 16);

  // workspace layout (floats)
  const size_t NTD = (size_t)NTROWS * DMODEL;
  float* ws    = (float*)d_ws;
  float* outb  = ws;                                  // activation [NT,256]
  float* posb  = outb + NTD;
  float* valb  = posb + NTD;
  float* vb    = valb + NTD;                          // vproj output
  float* offb  = vb + NTD;                            // [NT,256]
  float* awb   = offb + NTD;                          // [NT,128]
  float* msdab = awb + (size_t)NTROWS * 128;          // [NT,256]
  float* tmpb  = msdab + NTD;                         // [NT,256]
  float* ffnb  = tmpb + NTD;                          // [NT,1024]

  cis_prep_kernel<<<NTROWS, 256, 0, stream>>>(
      src[0], src[1], src[2], src[3], pos[0], pos[1], pos[2], pos[3],
      mem[0], mem[1], mem[2], mem[3], posm[0], posm[1], posm[2], posm[3],
      le, outb, posb, valb);

  auto gemm = [&](const float* A, const float* A2, const float* W, const float* bias,
                  float* C, int N, int K, int relu) {
    dim3 grid(NTROWS / BM, N / BN);
    cis_gemm_bf16_wmma<<<grid, 256, 0, stream>>>(A, A2, W, bias, C, NTROWS, N, K, relu);
  };

  for (int li = 0; li < NLAYERS; ++li) {
    int base = PB + li * 16;
    const float *awB = f(base + 0),  *awW = f(base + 1);
    const float *l1B = f(base + 2),  *l1W = f(base + 3);
    const float *l2B = f(base + 4),  *l2W = f(base + 5);
    const float *n1b = f(base + 6),  *n1g = f(base + 7);
    const float *n2b = f(base + 8),  *n2g = f(base + 9);
    const float *ofB = f(base + 10), *ofW = f(base + 11);
    const float *opB = f(base + 12), *opW = f(base + 13);
    const float *vpB = f(base + 14), *vpW = f(base + 15);

    // v = val @ Wv + b
    gemm(valb, nullptr, vpW, vpB, vb, DMODEL, DMODEL, 0);
    // q = out + pos (fused as A2): sampling offsets + attention weights
    gemm(outb, posb, ofW, ofB, offb, DMODEL, DMODEL, 0);
    gemm(outb, posb, awW, awB, awb, 128, DMODEL, 0);
    cis_softmax16<<<(NTROWS * NHEAD + 255) / 256, 256, 0, stream>>>(awb, NTROWS * NHEAD);
    cis_msda_sample<<<NTROWS * NHEAD / 8, 256, 0, stream>>>(vb, offb, awb, msdab);
    // oproj + residual + LN1
    gemm(msdab, nullptr, opW, opB, tmpb, DMODEL, DMODEL, 0);
    cis_add_ln<<<NTROWS / 8, 256, 0, stream>>>(outb, tmpb, n1g, n1b, outb, NTROWS);
    // FFN + residual + LN2
    gemm(outb, nullptr, l1W, l1B, ffnb, NFFN, DMODEL, 1);
    gemm(ffnb, nullptr, l2W, l2B, tmpb, DMODEL, NFFN, 0);
    cis_add_ln<<<NTROWS / 8, 256, 0, stream>>>(outb, tmpb, n2g, n2b, outb, NTROWS);
  }

  cis_finalize<<<(int)((NTD + 4 + 255) / 256), 256, 0, stream>>>(outb, (float*)d_out);
}
